// Attention_4956392259713
// MI455X (gfx1250) — matrix operand order repaired and checked by reading, not compiled, not run on hardware
//
#include <hip/hip_runtime.h>

typedef __attribute__((ext_vector_type(16))) _Float16 v16h;
typedef __attribute__((ext_vector_type(8)))  _Float16 v8h;
typedef __attribute__((ext_vector_type(8)))  float    v8f;
typedef __attribute__((ext_vector_type(4)))  float    v4f;

#define BATCH 8
#define SEQ   2048
#define HID   768
#define HID3  2304

// ---- monotone float<->uint encoding so row-max can use uint atomicMax ----
static __device__ __forceinline__ unsigned encf(float f) {
    unsigned b = __float_as_uint(f);
    return (b & 0x80000000u) ? ~b : (b | 0x80000000u);
}
static __device__ __forceinline__ float decf(unsigned u) {
    return (u & 0x80000000u) ? __uint_as_float(u & 0x7fffffffu)
                             : __uint_as_float(~u);
}

static __device__ __forceinline__ v16h combine16(v8h lo, v8h hi) {
    v16h a;
#pragma unroll
    for (int i = 0; i < 8; ++i) { a[i] = lo[i]; a[i + 8] = hi[i]; }
    return a;
}

// ---- CDNA5 async global->LDS copy (ASYNCcnt-tracked, no VGPR return) ----
// LDS operand: generic LDS pointers carry the LDS byte offset in addr[31:0].
static __device__ __forceinline__ void async_b128(const void* lds_generic,
                                                  const void* gaddr) {
    unsigned ldsoff = (unsigned)(size_t)lds_generic;
    asm volatile("global_load_async_to_lds_b128 %0, %1, off"
                 :: "v"(ldsoff), "v"(gaddr)
                 : "memory");
}
static __device__ __forceinline__ void wait_async2() {
    asm volatile("s_wait_asynccnt 0x2" ::: "memory");
}
static __device__ __forceinline__ void wait_async0() {
    asm volatile("s_wait_asynccnt 0x0" ::: "memory");
}

// B tile in LDS: 128 rows x 32 f16, row stride 40 halfs (80B) => conflict-free.
#define BSTRIDE 40

// ---------------- K0: convert W to f16, init row-max ----------------
__global__ __launch_bounds__(256) void prep_kernel(const float* __restrict__ W,
                                                   _Float16* __restrict__ W16,
                                                   unsigned* __restrict__ rowmaxU) {
    int idx = blockIdx.x * 256 + threadIdx.x;
    if (idx < HID3 * HID) W16[idx] = (_Float16)W[idx];
    if (idx < BATCH * SEQ) rowmaxU[idx] = encf(-3.0e38f);
}

// ---------------- K1: qkv = x @ W^T + b  (writes q16, k16, vT16) ----------------
// grid (18, 128); block 256 = 8 waves as 4(M)x2(N). Block tile 128 x 128.
__global__ __launch_bounds__(256) void qkv_kernel(const float* __restrict__ x,
                                                  const _Float16* __restrict__ W16,
                                                  const float* __restrict__ bias,
                                                  _Float16* __restrict__ q16,
                                                  _Float16* __restrict__ k16,
                                                  _Float16* __restrict__ vT16) {
    __shared__ __align__(16) _Float16 ldsA[128 * 40];          // x tile 128x32 f16
    __shared__ __align__(16) _Float16 ldsW[2][128 * BSTRIDE];  // W tile 128x32 f16, dbl buf

    const int tid  = threadIdx.x;
    const int lane = tid & 31, wv = tid >> 5;
    const int hf   = lane >> 4, l16 = lane & 15;
    const int wm   = wv >> 1, wn = wv & 1;
    const int nbase = blockIdx.x * 128;  // column in [0, 2304)
    const int mbase = blockIdx.y * 128;  // row    in [0, 16384)
    const int sel   = nbase / HID;       // 0=q 1=k 2=v  (768 % 128 == 0: no straddle)
    const int col0  = nbase - sel * HID;

    const int lr = tid >> 1;  // x loader row 0..127
    const int lh = tid & 1;   // x loader k-half (16 floats)
    const int brow = tid >> 1, bseg0 = (tid & 1) * 2;  // B loader: 128 rows x 2x16B each
    const _Float16* wbase = W16 + (size_t)(nbase + brow) * HID + bseg0 * 8;
    _Float16* const ldsWr0 = &ldsW[0][brow * BSTRIDE + bseg0 * 8];
    _Float16* const ldsWr1 = &ldsW[1][brow * BSTRIDE + bseg0 * 8];

    v8f acc[2][4] = {};

    async_b128(ldsWr0, wbase);
    async_b128(ldsWr0 + 8, wbase + 8);

    const int nIter = HID / 32;
    for (int kk = 0; kk < nIter; ++kk) {
        const int kblk = kk * 32;
        if (kk + 1 < nIter) {
            _Float16* dst = ((kk + 1) & 1) ? ldsWr1 : ldsWr0;
            async_b128(dst, wbase + (kblk + 32));
            async_b128(dst + 8, wbase + (kblk + 40));
        }
        // stage x tile (128 x 32) fp32 -> f16 into LDS
        {
            const float* src = x + (size_t)(mbase + lr) * HID + kblk + lh * 16;
            v4f f0 = *(const v4f*)(src + 0);
            v4f f1 = *(const v4f*)(src + 4);
            v4f f2 = *(const v4f*)(src + 8);
            v4f f3 = *(const v4f*)(src + 12);
            v8h h0, h1;
#pragma unroll
            for (int c = 0; c < 4; ++c) {
                h0[c] = (_Float16)f0[c]; h0[c + 4] = (_Float16)f1[c];
                h1[c] = (_Float16)f2[c]; h1[c + 4] = (_Float16)f3[c];
            }
            *(v8h*)&ldsA[lr * 40 + lh * 16]     = h0;
            *(v8h*)&ldsA[lr * 40 + lh * 16 + 8] = h1;
        }
        if (kk + 1 < nIter) wait_async2(); else wait_async0();
        __syncthreads();

        v16h af[2];
#pragma unroll
        for (int sub = 0; sub < 2; ++sub) {
            const int row = wm * 32 + sub * 16 + l16;
            v8h lo = *(v8h*)&ldsA[row * 40 + hf * 8];
            v8h hi = *(v8h*)&ldsA[row * 40 + 16 + hf * 8];
            af[sub] = combine16(lo, hi);
        }
        const _Float16* wb = &ldsW[kk & 1][0];
        v16h bfr[4];
#pragma unroll
        for (int t = 0; t < 4; ++t) {
            const int rb = wn * 64 + t * 16 + l16;
            v8h blo = *(v8h*)&wb[rb * BSTRIDE + hf * 8];
            v8h bhi = *(v8h*)&wb[rb * BSTRIDE + 16 + hf * 8];
            bfr[t] = combine16(blo, bhi);
        }
#pragma unroll
        for (int sub = 0; sub < 2; ++sub)
#pragma unroll
            for (int t = 0; t < 4; ++t)
                acc[sub][t] = __builtin_amdgcn_wmma_f32_16x16x32_f16(
                    false, af[sub], false, bfr[t], (short)0, acc[sub][t], false, false);
        __syncthreads();
    }

    const int mrow0 = mbase + wm * 32;  // + sub*16 + r + 8*hf
    if (sel < 2) {
        _Float16* outp = (sel == 0) ? q16 : k16;
#pragma unroll
        for (int sub = 0; sub < 2; ++sub)
#pragma unroll
            for (int t = 0; t < 4; ++t) {
                const int n  = col0 + wn * 64 + t * 16 + l16;
                const float bb = bias[sel * HID + n];
#pragma unroll
                for (int r = 0; r < 8; ++r) {
                    const int m = mrow0 + sub * 16 + r + 8 * hf;
                    outp[(size_t)m * HID + n] = (_Float16)(acc[sub][t][r] + bb);
                }
            }
    } else {
        // v written transposed: vT[b][d][i]
        const int bz = mrow0 / SEQ;  // block never straddles batches (128 | 2048)
#pragma unroll
        for (int sub = 0; sub < 2; ++sub) {
            const int i0 = (mrow0 - bz * SEQ) + sub * 16 + 8 * hf;
#pragma unroll
            for (int t = 0; t < 4; ++t) {
                const int d  = col0 + wn * 64 + t * 16 + l16;
                const float bb = bias[2 * HID + d];
                v8h pk;
#pragma unroll
                for (int r = 0; r < 8; ++r) pk[r] = (_Float16)(acc[sub][t][r] + bb);
                *(v8h*)(vT16 + ((size_t)bz * HID + d) * SEQ + i0) = pk;
            }
        }
    }
}

// ---------------- K2: S = q @ k^T (f16 out) + row-max atomics ----------------
// grid (16, 16, 8); block tile 128(q rows) x 128(keys), contraction 768.
__global__ __launch_bounds__(256) void qk_kernel(const _Float16* __restrict__ q16,
                                                 const _Float16* __restrict__ k16,
                                                 _Float16* __restrict__ S16,
                                                 unsigned* __restrict__ rowmaxU) {
    __shared__ __align__(16) _Float16 ldsB[2][128 * BSTRIDE];

    const int tid = threadIdx.x, lane = tid & 31, wv = tid >> 5;
    const int hf = lane >> 4, l16 = lane & 15;
    const int wm = wv >> 1, wn = wv & 1;
    const int nb = blockIdx.x * 128;  // key base
    const int mb = blockIdx.y * 128;  // query base
    const int bz = blockIdx.z;

    const _Float16* qrow0 = q16 + ((size_t)bz * SEQ + mb + wm * 32 + l16) * HID;
    const _Float16* qrow1 = qrow0 + (size_t)16 * HID;

    const int brow = tid >> 1, bseg0 = (tid & 1) * 2;
    const _Float16* kbase = k16 + ((size_t)bz * SEQ + nb + brow) * HID + bseg0 * 8;
    _Float16* const ldsr0 = &ldsB[0][brow * BSTRIDE + bseg0 * 8];
    _Float16* const ldsr1 = &ldsB[1][brow * BSTRIDE + bseg0 * 8];

    v8f acc[2][4] = {};
    async_b128(ldsr0, kbase);
    async_b128(ldsr0 + 8, kbase + 8);

    const int nIter = HID / 32;
    for (int kk = 0; kk < nIter; ++kk) {
        const int kblk = kk * 32;
        if (kk + 1 < nIter) {
            _Float16* dst = ((kk + 1) & 1) ? ldsr1 : ldsr0;
            async_b128(dst, kbase + (kblk + 32));
            async_b128(dst + 8, kbase + (kblk + 40));
        }
        v16h af[2];
        af[0] = combine16(*(const v8h*)(qrow0 + kblk + hf * 8),
                          *(const v8h*)(qrow0 + kblk + 16 + hf * 8));
        af[1] = combine16(*(const v8h*)(qrow1 + kblk + hf * 8),
                          *(const v8h*)(qrow1 + kblk + 16 + hf * 8));
        if (kk + 1 < nIter) wait_async2(); else wait_async0();
        __syncthreads();
        const _Float16* bb = &ldsB[kk & 1][0];
        v16h bfr[4];
#pragma unroll
        for (int t = 0; t < 4; ++t) {
            const int rb = wn * 64 + t * 16 + l16;
            v8h blo = *(v8h*)&bb[rb * BSTRIDE + hf * 8];
            v8h bhi = *(v8h*)&bb[rb * BSTRIDE + 16 + hf * 8];
            bfr[t] = combine16(blo, bhi);
        }
#pragma unroll
        for (int sub = 0; sub < 2; ++sub)
#pragma unroll
            for (int t = 0; t < 4; ++t)
                acc[sub][t] = __builtin_amdgcn_wmma_f32_16x16x32_f16(
                    false, af[sub], false, bfr[t], (short)0, acc[sub][t], false, false);
        __syncthreads();
    }

    // per-row max over this 128-wide tile, reduced across the 16-lane half-wave
#pragma unroll
    for (int sub = 0; sub < 2; ++sub) {
        float rmax[8];
#pragma unroll
        for (int r = 0; r < 8; ++r)
            rmax[r] = fmaxf(fmaxf(acc[sub][0][r], acc[sub][1][r]),
                            fmaxf(acc[sub][2][r], acc[sub][3][r]));
#pragma unroll
        for (int mask = 1; mask < 16; mask <<= 1)
#pragma unroll
            for (int r = 0; r < 8; ++r)
                rmax[r] = fmaxf(rmax[r], __shfl_xor(rmax[r], mask, 32));
        if (l16 == 0) {
#pragma unroll
            for (int r = 0; r < 8; ++r) {
                const int i = mb + wm * 32 + sub * 16 + r + 8 * hf;
                atomicMax(&rowmaxU[bz * SEQ + i], encf(rmax[r]));
            }
        }
    }
#pragma unroll
    for (int sub = 0; sub < 2; ++sub)
#pragma unroll
        for (int t = 0; t < 4; ++t) {
            const int j = nb + wn * 64 + t * 16 + l16;
#pragma unroll
            for (int r = 0; r < 8; ++r) {
                const int i = mb + wm * 32 + sub * 16 + r + 8 * hf;
                S16[((size_t)bz * SEQ + i) * SEQ + j] = (_Float16)acc[sub][t][r];
            }
        }
}

// ---------------- K3: P = exp(S - rowmax) in place, rowsum exact ----------------
__global__ __launch_bounds__(256) void softmax_kernel(_Float16* __restrict__ S16,
                                                      const unsigned* __restrict__ rowmaxU,
                                                      float* __restrict__ rowsum) {
    __shared__ float wsum[8];
    const int row = blockIdx.x;
    const int tid = threadIdx.x;
    _Float16* srow = S16 + (size_t)row * SEQ;
    const float m = decf(rowmaxU[row]);

    v8h vals = *(v8h*)(srow + tid * 8);
    v8h outp;
    float s = 0.f;
#pragma unroll
    for (int c = 0; c < 8; ++c) {
        float p = __expf((float)vals[c] - m);
        s += p;
        outp[c] = (_Float16)p;
    }
    *(v8h*)(srow + tid * 8) = outp;

#pragma unroll
    for (int mask = 1; mask < 32; mask <<= 1) s += __shfl_xor(s, mask, 32);
    if ((tid & 31) == 0) wsum[tid >> 5] = s;
    __syncthreads();
    if (tid < 8) {
        float t = wsum[tid];
#pragma unroll
        for (int mask = 1; mask < 8; mask <<= 1) t += __shfl_xor(t, mask, 32);
        if (tid == 0) rowsum[row] = t;
    }
}

// ---------------- K4: O = (P @ vT^T) / rowsum ----------------
// grid (6, 16, 8); block tile 128(q rows) x 128(d cols), contraction 2048.
__global__ __launch_bounds__(256) void pv_kernel(const _Float16* __restrict__ P16,
                                                 const _Float16* __restrict__ vT16,
                                                 const float* __restrict__ rowsum,
                                                 float* __restrict__ out) {
    __shared__ __align__(16) _Float16 ldsB[2][128 * BSTRIDE];

    const int tid = threadIdx.x, lane = tid & 31, wv = tid >> 5;
    const int hf = lane >> 4, l16 = lane & 15;
    const int wm = wv >> 1, wn = wv & 1;
    const int db = blockIdx.x * 128;
    const int mb = blockIdx.y * 128;
    const int bz = blockIdx.z;

    const _Float16* prow0 = P16 + ((size_t)bz * SEQ + mb + wm * 32 + l16) * SEQ;
    const _Float16* prow1 = prow0 + (size_t)16 * SEQ;

    const int brow = tid >> 1, bseg0 = (tid & 1) * 2;
    const _Float16* vbase = vT16 + ((size_t)bz * HID + db + brow) * SEQ + bseg0 * 8;
    _Float16* const ldsr0 = &ldsB[0][brow * BSTRIDE + bseg0 * 8];
    _Float16* const ldsr1 = &ldsB[1][brow * BSTRIDE + bseg0 * 8];

    v8f acc[2][4] = {};
    async_b128(ldsr0, vbase);
    async_b128(ldsr0 + 8, vbase + 8);

    const int nIter = SEQ / 32;
    for (int kk = 0; kk < nIter; ++kk) {
        const int kblk = kk * 32;
        if (kk + 1 < nIter) {
            _Float16* dst = ((kk + 1) & 1) ? ldsr1 : ldsr0;
            async_b128(dst, vbase + (kblk + 32));
            async_b128(dst + 8, vbase + (kblk + 40));
        }
        v16h af[2];
        af[0] = combine16(*(const v8h*)(prow0 + kblk + hf * 8),
                          *(const v8h*)(prow0 + kblk + 16 + hf * 8));
        af[1] = combine16(*(const v8h*)(prow1 + kblk + hf * 8),
                          *(const v8h*)(prow1 + kblk + 16 + hf * 8));
        if (kk + 1 < nIter) wait_async2(); else wait_async0();
        __syncthreads();
        const _Float16* bb = &ldsB[kk & 1][0];
        v16h bfr[4];
#pragma unroll
        for (int t = 0; t < 4; ++t) {
            const int rb = wn * 64 + t * 16 + l16;
            v8h blo = *(v8h*)&bb[rb * BSTRIDE + hf * 8];
            v8h bhi = *(v8h*)&bb[rb * BSTRIDE + 16 + hf * 8];
            bfr[t] = combine16(blo, bhi);
        }
#pragma unroll
        for (int sub = 0; sub < 2; ++sub)
#pragma unroll
            for (int t = 0; t < 4; ++t)
                acc[sub][t] = __builtin_amdgcn_wmma_f32_16x16x32_f16(
                    false, af[sub], false, bfr[t], (short)0, acc[sub][t], false, false);
        __syncthreads();
    }

    float scl[2][8];
#pragma unroll
    for (int sub = 0; sub < 2; ++sub)
#pragma unroll
        for (int r = 0; r < 8; ++r)
            scl[sub][r] = 1.0f / rowsum[bz * SEQ + mb + wm * 32 + sub * 16 + r + 8 * hf];
#pragma unroll
    for (int sub = 0; sub < 2; ++sub)
#pragma unroll
        for (int t = 0; t < 4; ++t) {
            const int d = db + wn * 64 + t * 16 + l16;
#pragma unroll
            for (int r = 0; r < 8; ++r) {
                const int i = mb + wm * 32 + sub * 16 + r + 8 * hf;
                out[((size_t)bz * SEQ + i) * HID + d] = acc[sub][t][r] * scl[sub][r];
            }
        }
}

extern "C" void kernel_launch(void* const* d_in, const int* in_sizes, int n_in,
                              void* d_out, int out_size, void* d_ws, size_t ws_size,
                              hipStream_t stream) {
    const float* x    = (const float*)d_in[0];
    const float* W    = (const float*)d_in[1];
    const float* bias = (const float*)d_in[2];
    float* out        = (float*)d_out;

    char* ws = (char*)d_ws;
    size_t off = 0;
    auto alloc = [&](size_t bytes) -> void* {
        void* p = ws + off;
        off += (bytes + 255) & ~(size_t)255;
        return p;
    };
    _Float16* q16     = (_Float16*)alloc((size_t)BATCH * SEQ * HID * 2);
    _Float16* k16     = (_Float16*)alloc((size_t)BATCH * SEQ * HID * 2);
    _Float16* vT16    = (_Float16*)alloc((size_t)BATCH * HID * SEQ * 2);
    _Float16* W16     = (_Float16*)alloc((size_t)HID3 * HID * 2);
    _Float16* S16     = (_Float16*)alloc((size_t)BATCH * SEQ * SEQ * 2);
    unsigned* rowmaxU = (unsigned*)alloc((size_t)BATCH * SEQ * 4);
    float*    rowsum  = (float*)alloc((size_t)BATCH * SEQ * 4);
    if (off > ws_size) return;  // workspace too small: bail rather than corrupt memory

    prep_kernel<<<(HID3 * HID + 255) / 256, 256, 0, stream>>>(W, W16, rowmaxU);
    qkv_kernel<<<dim3(HID3 / 128, (BATCH * SEQ) / 128), 256, 0, stream>>>(x, W16, bias,
                                                                          q16, k16, vT16);
    qk_kernel<<<dim3(SEQ / 128, SEQ / 128, BATCH), 256, 0, stream>>>(q16, k16, S16, rowmaxU);
    softmax_kernel<<<BATCH * SEQ, 256, 0, stream>>>(S16, rowmaxU, rowsum);
    pv_kernel<<<dim3(HID / 128, SEQ / 128, BATCH), 256, 0, stream>>>(S16, vT16, rowsum, out);
}
